// CPCModule_67448166416942
// MI455X (gfx1250) — matrix operand order repaired and checked by reading, not compiled, not run on hardware
//
#include <hip/hip_runtime.h>
#include <math.h>

#define NB      16
#define LOUT    2047
#define NC      256
#define NT      10240
#define KSTEPS  12
#define NNEG    10
#define KER     10
#define STR     5
#define ROWS    (NB * LOUT)    // 32752
#define TILES   (ROWS / 16)    // 2047 (odd)
#define TILESP  2048           // padded tile count
#define ROWSP   (TILESP * 16)  // 32768
#define PAIRS   (TILESP / 2)   // 1024

typedef __attribute__((ext_vector_type(16))) _Float16 v16h;
typedef __attribute__((ext_vector_type(8)))  float    v8f;

// ---------------- Stage 1: conv + relu channel statistics -------------------
__global__ void k_conv_stats(const float* __restrict__ x, const float* __restrict__ cw,
                             const float* __restrict__ cb,
                             float* __restrict__ s1, float* __restrict__ s2) {
    int c = blockIdx.x;
    __shared__ float wsh[16];
    if (threadIdx.x < KER) wsh[threadIdx.x] = cw[c * KER + threadIdx.x];
    __syncthreads();
    float bias = cb[c];
    float a1 = 0.f, a2 = 0.f;
    for (int i = threadIdx.x; i < ROWS; i += blockDim.x) {
        int b = i / LOUT, l = i % LOUT;
        const float* xp = x + b * NT + l * STR;
        float v = 0.f;
#pragma unroll
        for (int kt = 0; kt < KER; ++kt) v += xp[kt] * wsh[kt];
        v = fmaxf(v + bias, 0.f);
        a1 += v; a2 += v * v;
    }
    __shared__ float r1[256], r2[256];
    r1[threadIdx.x] = a1; r2[threadIdx.x] = a2;
    __syncthreads();
    for (int s = 128; s > 0; s >>= 1) {
        if (threadIdx.x < s) { r1[threadIdx.x] += r1[threadIdx.x + s]; r2[threadIdx.x] += r2[threadIdx.x + s]; }
        __syncthreads();
    }
    if (threadIdx.x == 0) { s1[c] = r1[0]; s2[c] = r2[0]; }
}

// ---------------- Stage 2: BN scale/shift -----------------------------------
__global__ void k_bn_finalize(const float* __restrict__ s1, const float* __restrict__ s2,
                              const float* __restrict__ gamma, const float* __restrict__ beta,
                              float* __restrict__ scale, float* __restrict__ shift) {
    int c = threadIdx.x;
    float n  = (float)ROWS;
    float mu = s1[c] / n;
    float var = s2[c] / n - mu * mu;
    float sc = gamma[c] * rsqrtf(var + 1e-5f);
    scale[c] = sc;
    shift[c] = beta[c] - mu * sc;
}

// ---------------- Stage 3: recompute conv, normalize, write z (B,C,L) -------
__global__ void k_normalize(const float* __restrict__ x, const float* __restrict__ cw,
                            const float* __restrict__ cb,
                            const float* __restrict__ scale, const float* __restrict__ shift,
                            float* __restrict__ out) {
    int idx = blockIdx.x * blockDim.x + threadIdx.x;   // idx = (b*NC + c)*LOUT + l
    int l  = idx % LOUT;
    int bc = idx / LOUT;
    int c  = bc % NC;
    int b  = bc / NC;
    const float* xp = x + b * NT + l * STR;
    const float* wr = cw + c * KER;
    float v = 0.f;
#pragma unroll
    for (int kt = 0; kt < KER; ++kt) v += xp[kt] * wr[kt];
    v = fmaxf(v + cb[c], 0.f);
    out[idx] = v * scale[c] + shift[c];
}

// ---------------- Stage 4a: pack z / z_neg into WMMA A-operand layout -------
// A (16x32 f16) lane layout: lanes 0-15 (M=lane): halves 0..7 -> K 0..7,
// halves 8..15 -> K 16..23 ; lanes 16-31 (M=lane-16): halves 0..7 -> K 8..15,
// halves 8..15 -> K 24..31.  Packed index = ((t*8+kb)*32+lane)*16 + h.
// Tiles are padded to TILESP; pad rows clamp to the last real row.
__global__ void k_packA(const float* __restrict__ zout, const int* __restrict__ perm,
                        _Float16* __restrict__ z16, _Float16* __restrict__ zn16,
                        _Float16* __restrict__ zr16, _Float16* __restrict__ znr16) {
    int idx  = blockIdx.x * blockDim.x + threadIdx.x;
    int h    = idx & 15;
    int lane = (idx >> 4) & 31;
    int kb   = (idx >> 9) & 7;
    int t    = idx >> 12;
    int M    = lane & 15;
    int base = (lane < 16) ? 0 : 8;
    int Klocal = (h < 8) ? (base + h) : (16 + base + (h - 8));
    int ccol = kb * 32 + Klocal;
    int row  = t * 16 + M;
    int srow = (row < ROWS) ? row : (ROWS - 1);      // clamp pad tile
    int b = srow / LOUT, l = srow % LOUT;
    float vp = zout[(b * NC + ccol) * LOUT + l];
    int bn = (b + NB - 1) % NB;
    int ln = perm[l];
    float vn = zout[(bn * NC + ccol) * LOUT + ln];
    _Float16 hp = (_Float16)vp, hn = (_Float16)vn;
    z16[idx]  = hp;
    zn16[idx] = hn;
    zr16[(size_t)srow * NC + ccol]  = hp;    // duplicates in pad region write same value
    znr16[(size_t)srow * NC + ccol] = hn;
}

// ---------------- Stage 4b: pack W into WMMA B-operand layout ----------------
// B (32x16 f16): lanes 0-15 (N=lane): halves 0..7 -> K 0..7, 8..15 -> K 16..23 ; lanes 16-31: halves 0..7 -> K 8..15, 8..15 -> K 24..31.
// Packed index = (((k*8+kb)*16+j)*32+lane)*16 + h.
__global__ void k_packW(const float* __restrict__ trw, _Float16* __restrict__ w16) {
    int idx  = blockIdx.x * blockDim.x + threadIdx.x;
    int h    = idx & 15;
    int lane = (idx >> 4) & 31;
    int j    = (idx >> 9) & 15;
    int kb   = (idx >> 13) & 7;
    int k    = idx >> 16;
    int K = kb * 32 + ((h < 8) ? (8 * (lane >> 4) + h) : (16 + 8 * (lane >> 4) + (h - 8)));
    int N = j * 16 + (lane & 15);
    w16[idx] = (_Float16)trw[k * 65536 + K * 256 + N];
}

// ---------------- Stage 5: bilinear forms f = (zW + b) . z via WMMA ----------
// One wave handles TWO 16-row tiles: each B tile feeds 2 independent wmma,
// halving W traffic and removing back-to-back accumulator dependencies.
__global__ void __launch_bounds__(128)
k_bilinear(const _Float16* __restrict__ z16,  const _Float16* __restrict__ zn16,
           const _Float16* __restrict__ w16,
           const _Float16* __restrict__ zr16, const _Float16* __restrict__ znr16,
           const float* __restrict__ trb, float* __restrict__ pq) {
    int wave = (blockIdx.x * blockDim.x + threadIdx.x) >> 5;
    int lane = threadIdx.x & 31;
    int p  = wave % PAIRS;
    int km = wave / PAIRS;        // km = k*2 + m
    int m  = km & 1;
    int k  = km >> 1;
    int t0 = p * 2, t1 = t0 + 1;

    const _Float16* Zp = (m == 0 ? z16  : zn16);
    const _Float16* Zr = (m == 0 ? zr16 : znr16);
    const _Float16* Wk = w16 + (size_t)k * 65536;

    // Preload A operands for both tiles (K = 0..255): 128 VGPRs
    v16h a0[8], a1[8];
#pragma unroll
    for (int kb = 0; kb < 8; ++kb) {
        a0[kb] = *(const v16h*)(Zp + (size_t)t0 * 4096 + kb * 512 + lane * 16);
        a1[kb] = *(const v16h*)(Zp + (size_t)t1 * 4096 + kb * 512 + lane * 16);
    }

    int cl    = lane & 15;                   // N within 16-col block
    int half8 = (lane >= 16) ? 8 : 0;        // D: lanes<16 hold M=v, lanes>=16 M=v+8
    int rb0 = t0 * 16 + half8;
    int rb1 = t1 * 16 + half8;

    // clamped row-major offsets for diag reads (pad tiles clamp to last row)
    int off0[8], off1[8];
#pragma unroll
    for (int v = 0; v < 8; ++v) {
        int r0 = rb0 + v; if (r0 > ROWS - 1) r0 = ROWS - 1;
        int r1 = rb1 + v; if (r1 > ROWS - 1) r1 = ROWS - 1;
        off0[v] = r0 * NC;
        off1[v] = r1 * NC;
    }

    float part0[8] = {}, part1[8] = {};
    for (int j = 0; j < 16; ++j) {
        v8f acc0 = {}, acc1 = {};
#pragma unroll
        for (int kb = 0; kb < 8; ++kb) {
            v16h bt = *(const v16h*)(Wk + kb * 8192 + j * 512 + lane * 16);
            acc0 = __builtin_amdgcn_wmma_f32_16x16x32_f16(
                false, a0[kb], false, bt, (short)0, acc0, false, false);
            acc1 = __builtin_amdgcn_wmma_f32_16x16x32_f16(
                false, a1[kb], false, bt, (short)0, acc1, false, false);
        }
        int col = j * 16 + cl;
        float bias = trb[k * NC + col];
#pragma unroll
        for (int v = 0; v < 8; ++v) {
            part0[v] += (acc0[v] + bias) * (float)Zr[off0[v] + col];
            part1[v] += (acc1[v] + bias) * (float)Zr[off1[v] + col];
        }
    }

    // reduce over the 16 columns (lanes 0-15 and 16-31 reduce independently)
#pragma unroll
    for (int v = 0; v < 8; ++v) {
        float s0 = part0[v], s1v = part1[v];
        s0 += __shfl_xor(s0, 8, 32); s0 += __shfl_xor(s0, 4, 32);
        s0 += __shfl_xor(s0, 2, 32); s0 += __shfl_xor(s0, 1, 32);
        s1v += __shfl_xor(s1v, 8, 32); s1v += __shfl_xor(s1v, 4, 32);
        s1v += __shfl_xor(s1v, 2, 32); s1v += __shfl_xor(s1v, 1, 32);
        part0[v] = s0; part1[v] = s1v;
    }
    if (cl == 0) {
        float* d0 = pq + (size_t)km * ROWSP + rb0;
        float* d1 = pq + (size_t)km * ROWSP + rb1;
#pragma unroll
        for (int v = 0; v < 8; ++v) { d0[v] = part0[v]; d1[v] = part1[v]; }
    }
}

// ---------------- Stage 6: InfoNCE-style loss (sum of -logp over all classes)
__global__ void k_loss(const float* __restrict__ pq, float* __restrict__ out_losses) {
    int k = blockIdx.x;          // 0..11  -> step kstep = k+1
    int kstep = k + 1;
    const float* P = pq + (size_t)(2 * k) * ROWSP;      // positive stream
    const float* G = P + ROWSP;                         // negative stream
    int llen = LOUT - kstep;
    float acc = 0.f;
    for (int i = threadIdx.x; i < NB * llen; i += blockDim.x) {
        int b = i / llen;
        int l = kstep + (i % llen);
        float p  = P[b * LOUT + l];
        float mx = p;
        float g[NNEG];
#pragma unroll
        for (int j = 0; j < NNEG; ++j) {
            g[j] = G[((b - j + NB) % NB) * LOUT + l];
            mx = fmaxf(mx, g[j]);
        }
        float se = expf(p - mx);
        float fs = p;
#pragma unroll
        for (int j = 0; j < NNEG; ++j) { se += expf(g[j] - mx); fs += g[j]; }
        float lse = logf(se) + mx;
        // reference sums -logp over ALL (NEG+1) classes
        acc += (float)(NNEG + 1) * lse - fs;
    }
    __shared__ float red[256];
    red[threadIdx.x] = acc;
    __syncthreads();
    for (int s = 128; s > 0; s >>= 1) {
        if (threadIdx.x < s) red[threadIdx.x] += red[threadIdx.x + s];
        __syncthreads();
    }
    if (threadIdx.x == 0)
        out_losses[k] = red[0] / ((float)(LOUT - 2 * kstep) * (float)NB);
}

// ---------------------------------------------------------------------------
extern "C" void kernel_launch(void* const* d_in, const int* in_sizes, int n_in,
                              void* d_out, int out_size, void* d_ws, size_t ws_size,
                              hipStream_t stream) {
    const float* x      = (const float*)d_in[0];
    const float* conv_w = (const float*)d_in[1];
    const float* conv_b = (const float*)d_in[2];
    const float* gamma  = (const float*)d_in[3];
    const float* beta   = (const float*)d_in[4];
    const float* trw    = (const float*)d_in[5];
    const float* trb    = (const float*)d_in[6];
    const int*   perm   = (const int*)d_in[7];
    float* out = (float*)d_out;               // [B*C*L] z, then [12] losses

    char* ws = (char*)d_ws;
    float* s1    = (float*)ws;                // 256
    float* s2    = s1 + NC;                   // 256
    float* scale = s2 + NC;                   // 256
    float* shift = scale + NC;                // 256
    _Float16* z16   = (_Float16*)(ws + 4096);                  // TILESP*4096 halves (A-layout)
    _Float16* zn16  = z16  + (size_t)TILESP * 4096;            // A-layout, negatives
    _Float16* w16   = zn16 + (size_t)TILESP * 4096;            // 12*256*256 halves (B-layout)
    float*    pq    = (float*)(w16 + (size_t)KSTEPS * NC * NC);   // 24*ROWSP floats
    _Float16* zr16  = (_Float16*)(pq + (size_t)KSTEPS * 2 * ROWSP); // row-major f16 z
    _Float16* znr16 = zr16 + (size_t)ROWS * NC;                     // row-major f16 z_neg

    k_conv_stats <<<NC, 256, 0, stream>>>(x, conv_w, conv_b, s1, s2);
    k_bn_finalize<<<1, NC, 0, stream>>>(s1, s2, gamma, beta, scale, shift);
    k_normalize  <<<(ROWS * NC) / 256, 256, 0, stream>>>(x, conv_w, conv_b, scale, shift, out);
    k_packA      <<<(TILESP * 4096) / 256, 256, 0, stream>>>(out, perm, z16, zn16, zr16, znr16);
    k_packW      <<<(KSTEPS * NC * NC) / 256, 256, 0, stream>>>(trw, w16);
    k_bilinear   <<<(KSTEPS * 2 * PAIRS * 32) / 128, 128, 0, stream>>>(z16, zn16, w16, zr16, znr16, trb, pq);
    k_loss       <<<KSTEPS, 256, 0, stream>>>(pq, out + (size_t)ROWS * NC);
}
